// MultiHeadLinearAttention_71476845740776
// MI455X (gfx1250) — matrix operand order repaired and checked by reading, not compiled, not run on hardware
//
#include <hip/hip_runtime.h>

typedef __bf16 bf16_t;
typedef __attribute__((ext_vector_type(16))) __bf16 v16bf;
typedef __attribute__((ext_vector_type(8)))  __bf16 v8bf;
typedef __attribute__((ext_vector_type(8)))  float  v8f;

namespace {

constexpr int kB  = 4;
constexpr int kS  = 4096;
constexpr int kD  = 1024;
constexpr int kH  = 16;
constexpr int kDK = 64;
constexpr int kM  = kB * kS;     // 16384 rows for all big GEMMs
constexpr float kEPS = 1e-6f;

enum Epi { EPI_RAW = 0, EPI_GLU_PHI = 1, EPI_GLU_NOPHI = 2, EPI_OUT = 3 };

// ---------------------------------------------------------------------------
// fp32 -> bf16, 8 elements per thread (2x float4 load, 4x v_cvt_pk, b128 store)
// ---------------------------------------------------------------------------
__global__ void cvt_f32_to_bf16_v8(const float4* __restrict__ src,
                                   v8bf* __restrict__ dst, int n8) {
  int i = blockIdx.x * blockDim.x + threadIdx.x;
  if (i < n8) {
    const float4 a = src[2 * i];
    const float4 b = src[2 * i + 1];
    v8bf o;
    o[0] = (bf16_t)a.x; o[1] = (bf16_t)a.y; o[2] = (bf16_t)a.z; o[3] = (bf16_t)a.w;
    o[4] = (bf16_t)b.x; o[5] = (bf16_t)b.y; o[6] = (bf16_t)b.z; o[7] = (bf16_t)b.w;
    dst[i] = o;
  }
}

// ---------------------------------------------------------------------------
// WMMA GEMM: Out = A[M,K]_bf16 @ W[N,K]_bf16^T (+ bias), fused epilogues.
//   block = 256x128 output, 256 threads = 8 waves (wave32),
//   wave = 64x64 (4x4 tiles of 16x16), K-step 32 (v_wmma_f32_16x16x32_bf16)
//   16 wmma : 16 b128 loads per K-step per wave.
// A-fragment (ISA 7.12.2, 16-bit A 16x32): lane l: row = m0 + (l&15);
//   elems [0..7]  = K: k0 + (l>>4)*8 + i
//   elems [8..15] = K: k0 + 16 + (l>>4)*8 + i
// B-fragment (32x16): lane l reads W[n0+(l&15)] cols k0+(l>>4)*8 .. +7, k0+16+(l>>4)*8 .. +7
// ---------------------------------------------------------------------------
template <int EPI>
__global__ __launch_bounds__(256)
void gemm_wmma_kernel(const bf16_t* __restrict__ Aptr,
                      const bf16_t* __restrict__ Wptr,
                      const float* __restrict__ bias,
                      void* __restrict__ Outptr,
                      const float* __restrict__ tmp,
                      const float* __restrict__ mask,
                      int M, int N, int K) {
  const int lane  = threadIdx.x & 31;
  const int wid   = threadIdx.x >> 5;
  const int wm    = wid >> 1;                 // 0..3
  const int wn    = wid & 1;                  // 0..1
  const int m0    = blockIdx.y * 256 + wm * 64;
  const int n0    = blockIdx.x * 128 + wn * 64;
  const int lrow  = lane & 15;
  const int khalf = lane >> 4;

  v8f acc[4][4];
#pragma unroll
  for (int mt = 0; mt < 4; ++mt)
#pragma unroll
    for (int nt = 0; nt < 4; ++nt) acc[mt][nt] = v8f{};

  for (int k0 = 0; k0 < K; k0 += 32) {
    // speculative prefetch of next K-slab of W (global_prefetch_b8, branchless)
    __builtin_prefetch(Wptr + (size_t)(n0 + lrow) * K + k0 + 32, 0, 1);

    v16bf afrag[4];
#pragma unroll
    for (int mt = 0; mt < 4; ++mt) {
      const bf16_t* p = Aptr + (size_t)(m0 + mt * 16 + lrow) * K + k0 + khalf * 8;
      const v8bf lo = *(const v8bf*)p;
      const v8bf hi = *(const v8bf*)(p + 16);
#pragma unroll
      for (int i = 0; i < 8; ++i) {
        afrag[mt][i]     = lo[i];
        afrag[mt][8 + i] = hi[i];
      }
    }

    v16bf bfrag[4];
#pragma unroll
    for (int nt = 0; nt < 4; ++nt) {
      const int n = n0 + nt * 16 + lrow;
      const bf16_t* q = Wptr + (size_t)n * K + k0 + khalf * 8;
      const v8bf blo = *(const v8bf*)q;
      const v8bf bhi = *(const v8bf*)(q + 16);
#pragma unroll
      for (int i = 0; i < 8; ++i) {
        bfrag[nt][i]     = blo[i];
        bfrag[nt][8 + i] = bhi[i];
      }
    }

#pragma unroll
    for (int mt = 0; mt < 4; ++mt)
#pragma unroll
      for (int nt = 0; nt < 4; ++nt)
        acc[mt][nt] = __builtin_amdgcn_wmma_f32_16x16x32_bf16(
            false, afrag[mt], false, bfrag[nt], (short)0, acc[mt][nt],
            false, false);
  }

  // Epilogue. C/D layout: VGPR r, lane l -> row tile + r + (l>>4)*8,
  // col tile + (l&15).
#pragma unroll
  for (int mt = 0; mt < 4; ++mt)
#pragma unroll
    for (int nt = 0; nt < 4; ++nt) {
      const int ncol = n0 + nt * 16 + lrow;
      const float bv = bias[ncol];
#pragma unroll
      for (int r = 0; r < 8; ++r) {
        const int m = m0 + mt * 16 + khalf * 8 + r;
        const size_t idx = (size_t)m * N + ncol;
        float v = acc[mt][nt][r] + bv;
        if constexpr (EPI == EPI_RAW || EPI == EPI_OUT) {
          ((float*)Outptr)[idx] = v;
        } else {
          const float z1 = tmp[idx];                       // x@W1^T + b1
          const float sg = z1 / (1.f + __expf(-z1));       // silu
          float val = sg * v;                              // * (x@W2^T + b2)
          if constexpr (EPI == EPI_GLU_PHI) {
            val = (val > 0.f) ? (val + 1.f) : __expf(val); // elu(x)+1
            if (mask) val *= mask[m];                      // phi_k * mask
          }
          ((bf16_t*)Outptr)[idx] = (bf16_t)val;
        }
      }
    }
}

// ---------------------------------------------------------------------------
// kv[b,h,dk,dv] = sum_s phi_k[b,s,h,dk] * v[b,s,h,dv];  ksum[b,h,dk]
// One workgroup per (b,h); tile S in chunks of 32 rows through LDS.
// ---------------------------------------------------------------------------
__global__ __launch_bounds__(256)
void kv_reduce_kernel(const bf16_t* __restrict__ phik,
                      const bf16_t* __restrict__ vbuf,
                      float* __restrict__ kv, float* __restrict__ ksum) {
  const int bh = blockIdx.x;
  const int b  = bh >> 4;
  const int h  = bh & 15;
  __shared__ float sk[32][64];
  __shared__ float sv[32][64];
  const int tid = threadIdx.x;
  const int dk  = tid >> 2;          // 0..63
  const int dvb = (tid & 3) << 4;    // 0,16,32,48

  float acc[16];
#pragma unroll
  for (int j = 0; j < 16; ++j) acc[j] = 0.f;
  float ks = 0.f;

  for (int s0 = 0; s0 < kS; s0 += 32) {
    for (int i = tid; i < 32 * 64; i += 256) {
      const int r = i >> 6, c = i & 63;
      const size_t g = ((size_t)(b * kS + s0 + r)) * kD + h * kDK + c;
      sk[r][c] = (float)phik[g];
      sv[r][c] = (float)vbuf[g];
    }
    __syncthreads();
#pragma unroll 4
    for (int r = 0; r < 32; ++r) {
      const float a = sk[r][dk];
#pragma unroll
      for (int j = 0; j < 16; ++j) acc[j] += a * sv[r][dvb + j];
    }
    if (tid < 64) {
#pragma unroll 4
      for (int r = 0; r < 32; ++r) ks += sk[r][tid];
    }
    __syncthreads();
  }
#pragma unroll
  for (int j = 0; j < 16; ++j)
    kv[((size_t)bh * kDK + dk) * kDK + dvb + j] = acc[j];
  if (tid < 64) ksum[bh * kDK + tid] = ks;
}

// ---------------------------------------------------------------------------
// attn[b,s,h,dv] = (phi_q[b,s,h,:] @ kv[b,h]) / (phi_q . ksum + eps)
// Grid (B*H, S/256); one thread per s-row; kv tile in LDS (16 KB).
// ---------------------------------------------------------------------------
__global__ __launch_bounds__(256)
void attn_apply_kernel(const bf16_t* __restrict__ phiq,
                       const float* __restrict__ kv,
                       const float* __restrict__ ksum,
                       bf16_t* __restrict__ attn) {
  const int bh = blockIdx.x;
  const int b  = bh >> 4;
  const int h  = bh & 15;
  __shared__ float skv[kDK * kDK];
  __shared__ float sks[kDK];
  const int tid = threadIdx.x;
  for (int i = tid; i < kDK * kDK; i += 256) skv[i] = kv[(size_t)bh * kDK * kDK + i];
  if (tid < kDK) sks[tid] = ksum[bh * kDK + tid];
  __syncthreads();

  const int s = blockIdx.y * 256 + tid;
  const size_t base = ((size_t)(b * kS + s)) * kD + h * kDK;

  float q[kDK];
#pragma unroll
  for (int k = 0; k < kDK; ++k) q[k] = (float)phiq[base + k];

  float z = 0.f;
#pragma unroll
  for (int k = 0; k < kDK; ++k) z += q[k] * sks[k];

  float num[kDK];
#pragma unroll
  for (int d = 0; d < kDK; ++d) num[d] = 0.f;
#pragma unroll 8
  for (int k = 0; k < kDK; ++k) {
    const float a = q[k];
#pragma unroll
    for (int d = 0; d < kDK; ++d) num[d] += a * skv[k * kDK + d];
  }
  const float inv = 1.f / (z + kEPS);
#pragma unroll
  for (int d = 0; d < kDK; ++d) attn[base + d] = (bf16_t)(num[d] * inv);
}

}  // namespace

// ---------------------------------------------------------------------------
extern "C" void kernel_launch(void* const* d_in, const int* in_sizes, int n_in,
                              void* d_out, int out_size, void* d_ws,
                              size_t ws_size, hipStream_t stream) {
  (void)in_sizes; (void)n_in; (void)out_size; (void)ws_size;

  const float* query = (const float*)d_in[0];
  const float* keyx  = (const float*)d_in[1];
  const float* value = (const float*)d_in[2];
  const float* mask  = (const float*)d_in[3];
  const float* fw[7] = {(const float*)d_in[4],  (const float*)d_in[6],   // q_w1,q_w2
                        (const float*)d_in[8],  (const float*)d_in[10],  // k_w1,k_w2
                        (const float*)d_in[12], (const float*)d_in[14],  // v_w1,v_w2
                        (const float*)d_in[16]};                         // out_w
  const float* b_q1 = (const float*)d_in[5];
  const float* b_q2 = (const float*)d_in[7];
  const float* b_k1 = (const float*)d_in[9];
  const float* b_k2 = (const float*)d_in[11];
  const float* b_v1 = (const float*)d_in[13];
  const float* b_v2 = (const float*)d_in[15];
  const float* b_o  = (const float*)d_in[17];

  // workspace layout
  char* ws = (char*)d_ws;
  const size_t nW = (size_t)kD * kD;               // 1M elems per weight
  const size_t nX = (size_t)kM * kD;               // 16.7M elems per activation
  bf16_t* wbf[7];
  size_t off = 0;
  for (int i = 0; i < 7; ++i) { wbf[i] = (bf16_t*)(ws + off); off += nW * sizeof(bf16_t); }
  bf16_t* xbf  = (bf16_t*)(ws + off); off += nX * sizeof(bf16_t);   // reused q->k->v
  float*  tmp  = (float*)(ws + off);  off += nX * sizeof(float);
  bf16_t* phiq = (bf16_t*)(ws + off); off += nX * sizeof(bf16_t);
  bf16_t* phik = (bf16_t*)(ws + off); off += nX * sizeof(bf16_t);
  bf16_t* vact = (bf16_t*)(ws + off); off += nX * sizeof(bf16_t);
  bf16_t* attn = (bf16_t*)(ws + off); off += nX * sizeof(bf16_t);
  float*  kvb  = (float*)(ws + off);  off += (size_t)kB * kH * kDK * kDK * sizeof(float);
  float*  ksum = (float*)(ws + off);  off += (size_t)kB * kH * kDK * sizeof(float);

  // 1) weights fp32 -> bf16
  const int gW = (int)((nW / 8 + 255) / 256);
  for (int i = 0; i < 7; ++i)
    cvt_f32_to_bf16_v8<<<gW, 256, 0, stream>>>((const float4*)fw[i],
                                               (v8bf*)wbf[i], (int)(nW / 8));

  const dim3 gG(kD / 128, kM / 256);  // (8, 64)
  const int  gX = (int)((nX / 8 + 255) / 256);

  // 2) GLU projections: z1 = x@W1^T+b1 (raw fp32), then fused
  //    phi(silu(z1)*(x@W2^T+b2)) written as bf16.  xbf reused serially.
  cvt_f32_to_bf16_v8<<<gX, 256, 0, stream>>>((const float4*)query, (v8bf*)xbf, (int)(nX / 8));
  gemm_wmma_kernel<EPI_RAW><<<gG, 256, 0, stream>>>(
      xbf, wbf[0], b_q1, tmp, nullptr, nullptr, kM, kD, kD);
  gemm_wmma_kernel<EPI_GLU_PHI><<<gG, 256, 0, stream>>>(
      xbf, wbf[1], b_q2, phiq, tmp, nullptr, kM, kD, kD);

  cvt_f32_to_bf16_v8<<<gX, 256, 0, stream>>>((const float4*)keyx, (v8bf*)xbf, (int)(nX / 8));
  gemm_wmma_kernel<EPI_RAW><<<gG, 256, 0, stream>>>(
      xbf, wbf[2], b_k1, tmp, nullptr, nullptr, kM, kD, kD);
  gemm_wmma_kernel<EPI_GLU_PHI><<<gG, 256, 0, stream>>>(
      xbf, wbf[3], b_k2, phik, tmp, mask, kM, kD, kD);

  cvt_f32_to_bf16_v8<<<gX, 256, 0, stream>>>((const float4*)value, (v8bf*)xbf, (int)(nX / 8));
  gemm_wmma_kernel<EPI_RAW><<<gG, 256, 0, stream>>>(
      xbf, wbf[4], b_v1, tmp, nullptr, nullptr, kM, kD, kD);
  gemm_wmma_kernel<EPI_GLU_NOPHI><<<gG, 256, 0, stream>>>(
      xbf, wbf[5], b_v2, vact, tmp, nullptr, kM, kD, kD);

  // 3) linear-attention summaries
  kv_reduce_kernel<<<kB * kH, 256, 0, stream>>>(phik, vact, kvb, ksum);
  attn_apply_kernel<<<dim3(kB * kH, kS / 256), 256, 0, stream>>>(phiq, kvb, ksum, attn);

  // 4) output projection (bf16 A) -> fp32 d_out
  gemm_wmma_kernel<EPI_OUT><<<gG, 256, 0, stream>>>(
      attn, wbf[6], b_o, d_out, nullptr, nullptr, kM, kD, kD);
}
